// AttentionGNNLSTM_51161650430199
// MI455X (gfx1250) — compile-verified, matrix operand order checked by reading, not run on hardware
//
#include <hip/hip_runtime.h>
#include <math.h>

// ---------------- model constants (match reference) ----------------
#define N_NODES 50000
#define N_EDGES 500000
#define E2T     (N_EDGES + N_NODES)   // edges + self loops
#define BATCH   64
#define SEQT    50
#define F_NODE_ 128
#define F_SEQ_  64
#define HID     64
#define H1_     4
#define NCLS    2
#define EMB     128                   // 2*HID
#define NHEAD   4
#define HD      32

typedef __attribute__((ext_vector_type(16))) _Float16 v16h;
typedef __attribute__((ext_vector_type(8)))  _Float16 v8h;
typedef __attribute__((ext_vector_type(8)))  float    v8f;

union F16x16 { v16h v; v8h h[2]; };

__device__ __forceinline__ float sigf(float x)  { return 1.f / (1.f + __expf(-x)); }
__device__ __forceinline__ float lrelu(float x) { return x > 0.f ? x : 0.2f * x; }

// float atomic max via monotonic int/uint orderings
__device__ __forceinline__ void atomicMaxF(float* a, float v) {
    if (v >= 0.f) atomicMax((int*)a, __float_as_int(v));
    else          atomicMin((unsigned int*)a, __float_as_uint(v));
}

// ---------------- utility kernels ----------------
__global__ void cvt_f16_k(const float* __restrict__ in, _Float16* __restrict__ out, int n) {
    int i = blockIdx.x * blockDim.x + threadIdx.x;
    if (i < n) out[i] = (_Float16)in[i];
}

__global__ void fill_f32_k(float* __restrict__ p, float v, int n) {
    int i = blockIdx.x * blockDim.x + threadIdx.x;
    if (i < n) p[i] = v;
}

// ---------------- generic WMMA GEMM: C[M,Nc] = A[M,K](f16) * W[Nc,K](f16)^T + bias ----------------
// One wave per 16x16 output tile. K multiple of 32, M,Nc multiples of 16.
__global__ void gemm_wmma_f16(const _Float16* __restrict__ A, int lda,
                              const _Float16* __restrict__ W, int ldw,
                              const float* __restrict__ bias,
                              float* __restrict__ C, int ldc,
                              int M, int Ncols, int K)
{
    int gwave = (int)((blockIdx.x * blockDim.x + threadIdx.x) >> 5);
    int lane  = threadIdx.x & 31;
    int ntiles = Ncols >> 4;
    int mt = gwave / ntiles;
    int nt = gwave - mt * ntiles;
    if (mt * 16 >= M) return;
    int l16  = lane & 15;
    int grp8 = (lane >> 4) << 3;        // lanes 16-31 hold K+8 .. per ISA A/B layout
    const _Float16* arow = A + (size_t)(mt * 16 + l16) * lda + grp8;
    const _Float16* brow = W + (size_t)(nt * 16 + l16) * ldw + grp8;
    v8f acc = {};
    for (int k0 = 0; k0 < K; k0 += 32) {
        F16x16 a, b;
        a.h[0] = *(const v8h*)(arow + k0);
        a.h[1] = *(const v8h*)(arow + k0 + 16);
        b.h[0] = *(const v8h*)(brow + k0);
        b.h[1] = *(const v8h*)(brow + k0 + 16);
        acc = __builtin_amdgcn_wmma_f32_16x16x32_f16(false, a.v, false, b.v,
                                                     (short)0, acc, false, false);
    }
    int n = nt * 16 + l16;
    float bv = bias ? bias[n] : 0.f;
    int mbase = mt * 16 + ((lane >> 4) << 3);
#pragma unroll
    for (int r = 0; r < 8; ++r)
        C[(size_t)(mbase + r) * ldc + n] = acc[r] + bv;
}

// ---------------- GAT helpers ----------------
// per-node attention dots: as_[n,h] = h1[n,h,:]·att_src[h,:]
__global__ void att_dot_k(const float* __restrict__ h, const float* __restrict__ att_s,
                          const float* __restrict__ att_d,
                          float* __restrict__ as_, float* __restrict__ ad_, int H)
{
    int i = blockIdx.x * blockDim.x + threadIdx.x;
    if (i >= N_NODES * H) return;
    int n = i / H, hh = i - n * H;
    const float* hp = h + (size_t)n * H * HID + hh * HID;
    const float* sp = att_s + hh * HID;
    const float* dp = att_d + hh * HID;
    float sa = 0.f, sd = 0.f;
    for (int c = 0; c < HID; ++c) { float v = hp[c]; sa += v * sp[c]; sd += v * dp[c]; }
    as_[i] = sa; ad_[i] = sd;
}

__device__ __forceinline__ void edge_sd(int e, const int* __restrict__ src,
                                        const int* __restrict__ dst, int& s, int& d) {
    if (e < N_EDGES) { s = src[e]; d = dst[e]; }
    else             { s = e - N_EDGES; d = s; }   // appended self loops
}

__global__ void edge_max_k(const int* __restrict__ src, const int* __restrict__ dst,
                           const float* __restrict__ as_, const float* __restrict__ ad_,
                           float* __restrict__ m, int H)
{
    int i = blockIdx.x * blockDim.x + threadIdx.x;
    if (i >= E2T * H) return;
    int e = i / H, hh = i - e * H;
    int s, d; edge_sd(e, src, dst, s, d);
    float v = lrelu(as_[s * H + hh] + ad_[d * H + hh]);
    atomicMaxF(&m[d * H + hh], v);
}

__global__ void edge_exp_k(const int* __restrict__ src, const int* __restrict__ dst,
                           const float* __restrict__ as_, const float* __restrict__ ad_,
                           const float* __restrict__ m, float* __restrict__ ee,
                           float* __restrict__ z, int H)
{
    int i = blockIdx.x * blockDim.x + threadIdx.x;
    if (i >= E2T * H) return;
    int e = i / H, hh = i - e * H;
    int s, d; edge_sd(e, src, dst, s, d);
    float v = lrelu(as_[s * H + hh] + ad_[d * H + hh]);
    float x = __expf(v - m[d * H + hh]);
    ee[i] = x;
    atomicAdd(&z[d * H + hh], x);
}

// heavy kernel: one wave per edge, lanes stride channels (coalesced gather + atomic scatter)
__global__ void edge_agg_k(const int* __restrict__ src, const int* __restrict__ dst,
                           const float* __restrict__ ee, const float* __restrict__ z,
                           const float* __restrict__ h, float* __restrict__ out, int H)
{
    int wv   = (int)((blockIdx.x * blockDim.x + threadIdx.x) >> 5);
    int lane = threadIdx.x & 31;
    if (wv >= E2T) return;
    int s, d; edge_sd(wv, src, dst, s, d);
    int C = H * HID;
    for (int ch = lane; ch < C; ch += 32) {
        int hh = ch >> 6;   // channel -> head (HID==64)
        float alpha = ee[(size_t)wv * H + hh] / z[d * H + hh];
        atomicAdd(&out[(size_t)d * C + ch], alpha * h[(size_t)s * C + ch]);
    }
}

__global__ void relu_bias_f16_k(const float* __restrict__ in, const float* __restrict__ b,
                                _Float16* __restrict__ out, int total, int cols)
{
    int i = blockIdx.x * blockDim.x + threadIdx.x;
    if (i >= total) return;
    int c = i % cols;
    out[i] = (_Float16)fmaxf(in[i] + b[c], 0.f);
}

__global__ void relu_bias_f32_k(float* __restrict__ io, const float* __restrict__ b,
                                int total, int cols)
{
    int i = blockIdx.x * blockDim.x + threadIdx.x;
    if (i >= total) return;
    int c = i % cols;
    io[i] = fmaxf(io[i] + b[c], 0.f);
}

// ---------------- graph mean pooling ----------------
__global__ void pool_sum_k(const float* __restrict__ g2, const int* __restrict__ batch,
                           float* __restrict__ sums, float* __restrict__ cnts)
{
    int wv   = (int)((blockIdx.x * blockDim.x + threadIdx.x) >> 5);
    int lane = threadIdx.x & 31;
    if (wv >= N_NODES) return;
    int b = batch[wv];
    for (int c = lane; c < HID; c += 32)
        atomicAdd(&sums[b * HID + c], g2[(size_t)wv * HID + c]);
    if (lane == 0) atomicAdd(&cnts[b], 1.f);
}

__global__ void pool_div_k(const float* __restrict__ sums, const float* __restrict__ cnts,
                           float* __restrict__ pooled)
{
    int i = blockIdx.x * blockDim.x + threadIdx.x;
    if (i >= BATCH * HID) return;
    pooled[i] = sums[i] / fmaxf(cnts[i / HID], 1.f);
}

// ---------------- bidirectional LSTM: one WG per direction, WMMA gates, h in LDS ----------------
__global__ __launch_bounds__(512) void lstm_wmma_k(
    const float* __restrict__ seq,             // [B,T,F_SEQ]
    const _Float16* __restrict__ WihF, const _Float16* __restrict__ WhhF,
    const float* __restrict__ bihF, const float* __restrict__ bhhF,
    const _Float16* __restrict__ WihB, const _Float16* __restrict__ WhhB,
    const float* __restrict__ bihB, const float* __restrict__ bhhB,
    _Float16* __restrict__ out16)              // [B,T,2*HID] f16
{
    const int dir = blockIdx.x;
    const _Float16* Wih = dir ? WihB : WihF;
    const _Float16* Whh = dir ? WhhB : WhhF;
    const float* bih = dir ? bihB : bihF;
    const float* bhh = dir ? bhhB : bhhF;

    __shared__ __align__(16) _Float16 sx[BATCH * HID];   // x_t (f16)
    __shared__ __align__(16) _Float16 sh[BATCH * HID];   // h_t (f16)

    const int tid  = threadIdx.x;
    const int lane = tid & 31;
    const int wave = tid >> 5;                 // 16 waves: 4 batch-tiles x 4 hidden-col tiles
    const int mrow = (wave >> 2) * 16;
    const int j0   = (wave & 3) * 16;
    const int l16  = lane & 15;
    const int grp8 = (lane >> 4) << 3;

    for (int i = tid; i < BATCH * HID; i += blockDim.x) sh[i] = (_Float16)0.f;

    v8f c = {};                                // cell state, resident in WMMA acc layout
    float bias[4];
#pragma unroll
    for (int q = 0; q < 4; ++q) { int n = q * HID + j0 + l16; bias[q] = bih[n] + bhh[n]; }

    for (int t = 0; t < SEQT; ++t) {
        int ts = dir ? (SEQT - 1 - t) : t;
        __syncthreads();                       // prior sh/sx uses done
        for (int i = tid; i < BATCH * HID; i += blockDim.x) {
            int b = i >> 6, f = i & 63;
            sx[i] = (_Float16)seq[((size_t)b * SEQT + ts) * F_SEQ_ + f];
        }
        __syncthreads();

        const _Float16* ax = sx + (mrow + l16) * HID + grp8;
        const _Float16* ah = sh + (mrow + l16) * HID + grp8;
        v8f acc[4] = {};
#pragma unroll
        for (int q = 0; q < 4; ++q) {
            int nrow = q * HID + j0 + l16;
            const _Float16* bx = Wih + nrow * HID + grp8;
            const _Float16* bhp = Whh + nrow * HID + grp8;
#pragma unroll
            for (int k0 = 0; k0 < HID; k0 += 32) {
                F16x16 a, b;
                a.h[0] = *(const v8h*)(ax + k0);  a.h[1] = *(const v8h*)(ax + k0 + 16);
                b.h[0] = *(const v8h*)(bx + k0);  b.h[1] = *(const v8h*)(bx + k0 + 16);
                acc[q] = __builtin_amdgcn_wmma_f32_16x16x32_f16(false, a.v, false, b.v,
                                                                (short)0, acc[q], false, false);
                a.h[0] = *(const v8h*)(ah + k0);  a.h[1] = *(const v8h*)(ah + k0 + 16);
                b.h[0] = *(const v8h*)(bhp + k0); b.h[1] = *(const v8h*)(bhp + k0 + 16);
                acc[q] = __builtin_amdgcn_wmma_f32_16x16x32_f16(false, a.v, false, b.v,
                                                                (short)0, acc[q], false, false);
            }
        }
        __syncthreads();                       // all sh reads complete before update

        int j = j0 + l16;
        int bbase = mrow + ((lane >> 4) << 3);
#pragma unroll
        for (int r = 0; r < 8; ++r) {
            float ig = sigf(acc[0][r] + bias[0]);
            float fg = sigf(acc[1][r] + bias[1]);
            float gg = tanhf(acc[2][r] + bias[2]);
            float og = sigf(acc[3][r] + bias[3]);
            c[r] = fg * c[r] + ig * gg;
            float hv = og * tanhf(c[r]);
            int bi = bbase + r;
            sh[bi * HID + j] = (_Float16)hv;
            out16[((size_t)bi * SEQT + ts) * (2 * HID) + dir * HID + j] = (_Float16)hv;
        }
    }
}

// ---------------- MHA core: one block per (batch, head) ----------------
__global__ void mha_k(const float* __restrict__ qkv, _Float16* __restrict__ o16)
{
    int bh = blockIdx.x;
    int b = bh >> 2, h = bh & 3;
    __shared__ float ks[SEQT * HD];
    __shared__ float vs[SEQT * HD];
    int tid = threadIdx.x;   // 64 threads
    for (int i = tid; i < SEQT * HD; i += blockDim.x) {
        int t = i / HD, d = i - t * HD;
        ks[i] = qkv[((size_t)b * SEQT + t) * 384 + 128 + h * HD + d];
        vs[i] = qkv[((size_t)b * SEQT + t) * 384 + 256 + h * HD + d];
    }
    __syncthreads();
    if (tid < SEQT) {
        float q[HD];
#pragma unroll
        for (int d = 0; d < HD; ++d) q[d] = qkv[((size_t)b * SEQT + tid) * 384 + h * HD + d];
        float sc[SEQT]; float mx = -1e30f;
        for (int j = 0; j < SEQT; ++j) {
            float s = 0.f;
#pragma unroll
            for (int d = 0; d < HD; ++d) s += q[d] * ks[j * HD + d];
            s *= 0.17677669529663687f;  // 1/sqrt(32)
            sc[j] = s; mx = fmaxf(mx, s);
        }
        float z = 0.f;
        for (int j = 0; j < SEQT; ++j) { sc[j] = __expf(sc[j] - mx); z += sc[j]; }
        float inv = 1.f / z;
        float o[HD] = {};
        for (int j = 0; j < SEQT; ++j) {
            float a = sc[j] * inv;
#pragma unroll
            for (int d = 0; d < HD; ++d) o[d] += a * vs[j * HD + d];
        }
#pragma unroll
        for (int d = 0; d < HD; ++d)
            o16[((size_t)b * SEQT + tid) * EMB + h * HD + d] = (_Float16)o[d];
    }
}

__global__ void attn_mean_k(const float* __restrict__ ao, float* __restrict__ pooled)
{
    int b = blockIdx.x, c = threadIdx.x;  // 128 threads
    float s = 0.f;
    for (int t = 0; t < SEQT; ++t) s += ao[((size_t)b * SEQT + t) * EMB + c];
    pooled[b * EMB + c] = s * (1.f / SEQT);
}

__global__ void final_fc_k(const float* __restrict__ gp, const float* __restrict__ ap,
                           const float* __restrict__ fcw, const float* __restrict__ fcb,
                           float* __restrict__ out)
{
    int i = blockIdx.x * blockDim.x + threadIdx.x;
    if (i >= BATCH * NCLS) return;
    int b = i / NCLS, k = i - b * NCLS;
    float s = fcb[k];
    for (int c = 0; c < HID; ++c) s += gp[b * HID + c] * fcw[k * (EMB + HID) + c];
    for (int c = 0; c < EMB; ++c) s += ap[b * EMB + c] * fcw[k * (EMB + HID) + HID + c];
    out[i] = s;
}

// ---------------- host launcher ----------------
static inline int cdiv(long long a, long long b) { return (int)((a + b - 1) / b); }

extern "C" void kernel_launch(void* const* d_in, const int* in_sizes, int n_in,
                              void* d_out, int out_size, void* d_ws, size_t ws_size,
                              hipStream_t stream)
{
    (void)in_sizes; (void)n_in; (void)out_size; (void)ws_size;
    const float* x     = (const float*)d_in[0];
    const int*   ei    = (const int*)d_in[1];
    const int*   batch = (const int*)d_in[2];
    const float* seq_x = (const float*)d_in[3];
    const float* g1W   = (const float*)d_in[4];
    const float* g1as  = (const float*)d_in[5];
    const float* g1ad  = (const float*)d_in[6];
    const float* g1b   = (const float*)d_in[7];
    const float* g2W   = (const float*)d_in[8];
    const float* g2as  = (const float*)d_in[9];
    const float* g2ad  = (const float*)d_in[10];
    const float* g2b   = (const float*)d_in[11];
    const float* WihF  = (const float*)d_in[12];
    const float* WhhF  = (const float*)d_in[13];
    const float* bihF  = (const float*)d_in[14];
    const float* bhhF  = (const float*)d_in[15];
    const float* WihB  = (const float*)d_in[16];
    const float* WhhB  = (const float*)d_in[17];
    const float* bihB  = (const float*)d_in[18];
    const float* bhhB  = (const float*)d_in[19];
    const float* Wqkv  = (const float*)d_in[20];
    const float* bqkv  = (const float*)d_in[21];
    const float* Wout  = (const float*)d_in[22];
    const float* bout  = (const float*)d_in[23];
    const float* fcw   = (const float*)d_in[24];
    const float* fcb   = (const float*)d_in[25];
    const int* srcp = ei;
    const int* dstp = ei + N_EDGES;

    char* ws = (char*)d_ws;
    size_t off = 0;
    auto alloc = [&](size_t bytes) -> char* {
        char* p = ws + off;
        off = (off + bytes + 255) & ~(size_t)255;
        return p;
    };
    _Float16* x16    = (_Float16*)alloc((size_t)N_NODES * F_NODE_ * 2);
    _Float16* W1h    = (_Float16*)alloc(256 * 128 * 2);
    _Float16* W2h    = (_Float16*)alloc(64 * 256 * 2);
    _Float16* WihF16 = (_Float16*)alloc(256 * 64 * 2);
    _Float16* WhhF16 = (_Float16*)alloc(256 * 64 * 2);
    _Float16* WihB16 = (_Float16*)alloc(256 * 64 * 2);
    _Float16* WhhB16 = (_Float16*)alloc(256 * 64 * 2);
    _Float16* Wq16   = (_Float16*)alloc(384 * 128 * 2);
    _Float16* Wo16   = (_Float16*)alloc(128 * 128 * 2);
    float* h1   = (float*)alloc((size_t)N_NODES * 256 * 4);
    float* as1  = (float*)alloc((size_t)N_NODES * 4 * 4);
    float* ad1  = (float*)alloc((size_t)N_NODES * 4 * 4);
    float* m1   = (float*)alloc((size_t)N_NODES * 4 * 4);
    float* z1   = (float*)alloc((size_t)N_NODES * 4 * 4);
    float* ee1  = (float*)alloc((size_t)E2T * 4 * 4);
    float* agg1 = (float*)alloc((size_t)N_NODES * 256 * 4);
    _Float16* g16 = (_Float16*)alloc((size_t)N_NODES * 256 * 2);
    float* h2   = (float*)alloc((size_t)N_NODES * 64 * 4);
    float* as2  = (float*)alloc((size_t)N_NODES * 4);
    float* ad2  = (float*)alloc((size_t)N_NODES * 4);
    float* m2   = (float*)alloc((size_t)N_NODES * 4);
    float* z2   = (float*)alloc((size_t)N_NODES * 4);
    float* ee2  = (float*)alloc((size_t)E2T * 4);
    float* agg2 = (float*)alloc((size_t)N_NODES * 64 * 4);
    float* sums = (float*)alloc(BATCH * HID * 4);
    float* cnts = (float*)alloc(BATCH * 4);
    float* gp   = (float*)alloc(BATCH * HID * 4);
    _Float16* lo16 = (_Float16*)alloc((size_t)BATCH * SEQT * EMB * 2);
    float* qkv  = (float*)alloc((size_t)BATCH * SEQT * 384 * 4);
    _Float16* o16 = (_Float16*)alloc((size_t)BATCH * SEQT * EMB * 2);
    float* ao   = (float*)alloc((size_t)BATCH * SEQT * EMB * 4);
    float* ap   = (float*)alloc(BATCH * EMB * 4);

    auto CVT = [&](const float* s, _Float16* d, int n) {
        cvt_f16_k<<<cdiv(n, 256), 256, 0, stream>>>(s, d, n);
    };
    CVT(x, x16, N_NODES * F_NODE_);
    CVT(g1W, W1h, 256 * 128);
    CVT(g2W, W2h, 64 * 256);
    CVT(WihF, WihF16, 256 * 64);  CVT(WhhF, WhhF16, 256 * 64);
    CVT(WihB, WihB16, 256 * 64);  CVT(WhhB, WhhB16, 256 * 64);
    CVT(Wqkv, Wq16, 384 * 128);
    CVT(Wout, Wo16, 128 * 128);

    auto FILL = [&](float* p, float v, long long n) {
        fill_f32_k<<<cdiv(n, 256), 256, 0, stream>>>(p, v, (int)n);
    };
    FILL(m1, -1e30f, (long long)N_NODES * 4);
    FILL(z1, 0.f,    (long long)N_NODES * 4);
    FILL(agg1, 0.f,  (long long)N_NODES * 256);
    FILL(m2, -1e30f, N_NODES);
    FILL(z2, 0.f,    N_NODES);
    FILL(agg2, 0.f,  (long long)N_NODES * 64);
    FILL(sums, 0.f,  BATCH * HID);
    FILL(cnts, 0.f,  BATCH);

    // --- GAT layer 1 ---
    {
        long long waves = (long long)(N_NODES / 16) * (256 / 16);
        gemm_wmma_f16<<<cdiv(waves * 32, 256), 256, 0, stream>>>(
            x16, 128, W1h, 128, nullptr, h1, 256, N_NODES, 256, 128);
    }
    att_dot_k<<<cdiv((long long)N_NODES * 4, 256), 256, 0, stream>>>(h1, g1as, g1ad, as1, ad1, 4);
    edge_max_k<<<cdiv((long long)E2T * 4, 256), 256, 0, stream>>>(srcp, dstp, as1, ad1, m1, 4);
    edge_exp_k<<<cdiv((long long)E2T * 4, 256), 256, 0, stream>>>(srcp, dstp, as1, ad1, m1, ee1, z1, 4);
    edge_agg_k<<<cdiv((long long)E2T * 32, 256), 256, 0, stream>>>(srcp, dstp, ee1, z1, h1, agg1, 4);
    relu_bias_f16_k<<<cdiv((long long)N_NODES * 256, 256), 256, 0, stream>>>(
        agg1, g1b, g16, N_NODES * 256, 256);

    // --- GAT layer 2 ---
    {
        long long waves = (long long)(N_NODES / 16) * (64 / 16);
        gemm_wmma_f16<<<cdiv(waves * 32, 256), 256, 0, stream>>>(
            g16, 256, W2h, 256, nullptr, h2, 64, N_NODES, 64, 256);
    }
    att_dot_k<<<cdiv(N_NODES, 256), 256, 0, stream>>>(h2, g2as, g2ad, as2, ad2, 1);
    edge_max_k<<<cdiv(E2T, 256), 256, 0, stream>>>(srcp, dstp, as2, ad2, m2, 1);
    edge_exp_k<<<cdiv(E2T, 256), 256, 0, stream>>>(srcp, dstp, as2, ad2, m2, ee2, z2, 1);
    edge_agg_k<<<cdiv((long long)E2T * 32, 256), 256, 0, stream>>>(srcp, dstp, ee2, z2, h2, agg2, 1);
    relu_bias_f32_k<<<cdiv((long long)N_NODES * 64, 256), 256, 0, stream>>>(
        agg2, g2b, N_NODES * 64, 64);

    // --- graph mean pool ---
    pool_sum_k<<<cdiv((long long)N_NODES * 32, 256), 256, 0, stream>>>(agg2, batch, sums, cnts);
    pool_div_k<<<cdiv(BATCH * HID, 256), 256, 0, stream>>>(sums, cnts, gp);

    // --- bidirectional LSTM (one WG per direction, WMMA gates) ---
    lstm_wmma_k<<<2, 512, 0, stream>>>(seq_x, WihF16, WhhF16, bihF, bhhF,
                                       WihB16, WhhB16, bihB, bhhB, lo16);

    // --- MHA ---
    {
        long long waves = (long long)(BATCH * SEQT / 16) * (384 / 16);
        gemm_wmma_f16<<<cdiv(waves * 32, 256), 256, 0, stream>>>(
            lo16, 128, Wq16, 128, bqkv, qkv, 384, BATCH * SEQT, 384, 128);
    }
    mha_k<<<BATCH * NHEAD, 64, 0, stream>>>(qkv, o16);
    {
        long long waves = (long long)(BATCH * SEQT / 16) * (128 / 16);
        gemm_wmma_f16<<<cdiv(waves * 32, 256), 256, 0, stream>>>(
            o16, 128, Wo16, 128, bout, ao, 128, BATCH * SEQT, 128, 128);
    }
    attn_mean_k<<<BATCH, EMB, 0, stream>>>(ao, ap);

    // --- head ---
    final_fc_k<<<1, 128, 0, stream>>>(gp, ap, fcw, fcb, (float*)d_out);
}
